// MultiHeadAttention_30399778521487
// MI455X (gfx1250) — matrix operand order repaired and checked by reading, not compiled, not run on hardware
//
#include <hip/hip_runtime.h>
#include <hip/hip_bf16.h>
#include <type_traits>
#include <math.h>

// ---------------------------------------------------------------------------
// Types for CDNA5 (gfx1250, wave32)
// ---------------------------------------------------------------------------
typedef __attribute__((ext_vector_type(16))) __bf16 bf16x16;
typedef __attribute__((ext_vector_type(8)))  __bf16 bf16x8;
typedef __attribute__((ext_vector_type(8)))  float  f32x8;
typedef __attribute__((ext_vector_type(4)))  unsigned int u32x4;
typedef __attribute__((ext_vector_type(8)))  int          i32x8;
typedef __attribute__((ext_vector_type(4)))  int          i32x4;

// Problem constants (match reference)
#define EMB_DIM   1024
#define QKV_DIM   1024
#define BATCH     8
#define SEQ       2048
#define BS_ROWS   (BATCH * SEQ)        // 16384
#define SM_SCALE  0.125f               // 1/sqrt(HEAD_DIM=64)

// GEMM tiling
#define BM 128
#define BN 128
#define BK 32
#define SK 40           // LDS row stride in bf16 (64B data + 16B pad -> matches TDM pad)
#define TILE_ELEMS (BM * SK)           // 5120 bf16 = 10240 B per tile

#if __has_builtin(__builtin_amdgcn_tensor_load_to_lds) && \
    __has_builtin(__builtin_amdgcn_s_wait_tensorcnt)
#define HAVE_TDM 1
#else
#define HAVE_TDM 0
#endif

// ---------------------------------------------------------------------------
// TDM tile load: copy a [128 rows x 32 elem] bf16 tile (row stride `ld` elems)
// from global memory into LDS at byte offset lds_off, inserting 16B of LDS
// padding after every 64B row (so LDS rows are SK=40 bf16 apart).
// D# per CDNA5 ISA ch.8: group0 = {ctl, lds_addr, global_addr, type},
// group1 = {flags/pad, tensor dims, tile dims, strides}. 2D => groups 2/3 zero.
// This toolchain exposes the 6-arg builtin form (g0, g1, g2, g3, g4, cpol).
// ---------------------------------------------------------------------------
#if HAVE_TDM
__device__ inline void tdm_load_tile(unsigned lds_off, const __bf16* gsrc, int ld) {
    unsigned long long ga = (unsigned long long)(uintptr_t)gsrc;
    u32x4 g0;
    g0[0] = 1u;                                   // count=1, user descriptor
    g0[1] = lds_off;                              // lds_addr (bytes)
    g0[2] = (unsigned)(ga & 0xffffffffull);       // global_addr[31:0]
    g0[3] = (unsigned)((ga >> 32) & 0x1ffffffull) // global_addr[56:32]
          | (2u << 30);                           // type=2 ("image")

    const unsigned td0 = (unsigned)ld;            // tensor_dim0 = row length
    const unsigned td1 = 0x00100000u;             // tensor_dim1: large (no OOB)
    const unsigned long long s0 = (unsigned long long)ld; // dim0 stride (elems)
    i32x8 g1;
    g1[0] = (1 << 16)                             // data_size = 2 bytes (bf16)
          | (1 << 20)                             // pad_enable
          | (3 << 22)                             // pad_interval: 16 DWORDs (64B)
          | (3 << 25);                            // pad_amount: 4 DWORDs (16B)
    g1[1] = (int)((td0 & 0xffffu) << 16);         // [15:0] atomic_barrier_addr=0
    g1[2] = (int)((td0 >> 16) | ((td1 & 0xffffu) << 16));
    g1[3] = (int)((td1 >> 16) | (32u << 16));     // tile_dim0 = 32 (K elems/row)
    g1[4] = (int)(128u);                          // tile_dim1 = 128 rows, dim2=0
    g1[5] = (int)(s0 & 0xffffffffull);
    g1[6] = (int)((s0 >> 32) & 0xffffull);        // dim1_stride = 0 (2D)
    g1[7] = 0;

    i32x4 z4 = {0, 0, 0, 0};
    i32x8 z8 = {0, 0, 0, 0, 0, 0, 0, 0};
    __builtin_amdgcn_tensor_load_to_lds(g0, g1, z4, z4, z8, 0);
}
#endif

// Fallback synchronous staging (bf16 only, [rows][K] layout)
__device__ inline void stage_rowsK_sync(const __bf16* __restrict__ src_base, int ld,
                                        __bf16* __restrict__ lds, int tid) {
    const int r = tid >> 1;
    const int c = (tid & 1) * 16;
    const bf16x8* s8 = reinterpret_cast<const bf16x8*>(src_base + (size_t)r * ld + c);
    __bf16* dst = lds + r * SK + c;
    *reinterpret_cast<bf16x8*>(dst)     = s8[0];
    *reinterpret_cast<bf16x8*>(dst + 8) = s8[1];
}

// ---------------------------------------------------------------------------
// Fragment loads from LDS ([row][K] tiles, stride SK bf16)
// A 16x32 frag (ISA 7.12.2): lane l<16 -> M=l, K in {0..7,16..23};
//                            lane>=16 -> M=l-16, K in {8..15,24..31}.
// ---------------------------------------------------------------------------
__device__ inline bf16x16 load_afrag(const __bf16* __restrict__ lds, int r0, int lane) {
    const int row = r0 + (lane & 15);
    const int kh  = (lane >> 4) * 8;
    bf16x8 lo = *reinterpret_cast<const bf16x8*>(lds + row * SK + kh);
    bf16x8 hi = *reinterpret_cast<const bf16x8*>(lds + row * SK + 16 + kh);
    bf16x16 f;
#pragma unroll
    for (int i = 0; i < 8; ++i) { f[i] = lo[i]; f[i + 8] = hi[i]; }
    return f;
}

// B 32x16 frag: lane l<16 -> N=l, K in {0..7,16..23}; lane>=16 -> N=l-16, K in {8..15,24..31}.
__device__ inline bf16x16 load_bfrag(const __bf16* __restrict__ lds, int r0, int lane) {
    const int row = r0 + (lane & 15);
    const int kb  = (lane >> 4) * 8;
    bf16x8 lo = *reinterpret_cast<const bf16x8*>(lds + row * SK + kb);
    bf16x8 hi = *reinterpret_cast<const bf16x8*>(lds + row * SK + kb + 16);
    bf16x16 f;
#pragma unroll
    for (int i = 0; i < 8; ++i) { f[i] = lo[i]; f[i + 8] = hi[i]; }
    return f;
}

// ---------------------------------------------------------------------------
// Uniform WMMA GEMM: C[M,N] = scale * A[M,K] x B[N,K]^T  (both bf16, [rows][K])
// CT=false: C[row*ldc + col]; CT=true: C[col*ldc + row] (transposed store).
// 256 threads = 8 waves (2x4 grid), each wave owns 64x32 (4x2 WMMA tiles).
// Double-buffered LDS; next tiles fetched by TDM (waves 0/1) during compute.
// ---------------------------------------------------------------------------
template <typename TC, bool CT>
__global__ __launch_bounds__(256)
void gemm_nt_wmma(const __bf16* __restrict__ A, const __bf16* __restrict__ B,
                  TC* __restrict__ C, int M, int N, int K,
                  int lda, int ldb, int ldc, float scale) {
    __shared__ __bf16 smem[4 * TILE_ELEMS];      // 2 buffers x (A tile + B tile)

    const int tid  = threadIdx.x;
    const int lane = tid & 31;
    const int wid  = tid >> 5;
    const int wr   = wid >> 2;                   // 0..1 -> 64-row slab
    const int wc   = wid & 3;                    // 0..3 -> 32-col slab
    const int m0   = blockIdx.y * BM;
    const int n0   = blockIdx.x * BN;

    const __bf16* Abase = A + (size_t)m0 * lda;
    const __bf16* Bbase = B + (size_t)n0 * ldb;

    f32x8 acc[4][2] = {};

#if HAVE_TDM
    const unsigned lds0 = (unsigned)(unsigned long long)(uintptr_t)smem;
    // Prologue: waves 0/1 DMA the k=0 tiles.
    if (wid == 0) {
        tdm_load_tile(lds0, Abase, lda);
        __builtin_amdgcn_s_wait_tensorcnt(0);
    } else if (wid == 1) {
        tdm_load_tile(lds0 + TILE_ELEMS * 2 * sizeof(__bf16), Bbase, ldb);
        __builtin_amdgcn_s_wait_tensorcnt(0);
    }
    __syncthreads();
#endif

    for (int k0 = 0; k0 < K; k0 += BK) {
        const int cur = (k0 / BK) & 1;
#if HAVE_TDM
        // Kick off DMA for the next K-step into the other buffer.
        if (k0 + BK < K) {
            const unsigned nb = lds0 + (unsigned)((cur ^ 1) * 2 * TILE_ELEMS * sizeof(__bf16));
            if (wid == 0)      tdm_load_tile(nb, Abase + (k0 + BK), lda);
            else if (wid == 1) tdm_load_tile(nb + TILE_ELEMS * sizeof(__bf16),
                                             Bbase + (k0 + BK), ldb);
        }
        const __bf16* As = smem + cur * 2 * TILE_ELEMS;
        const __bf16* Bs = As + TILE_ELEMS;
#else
        (void)cur;
        __bf16* As = smem;
        __bf16* Bs = smem + TILE_ELEMS;
        stage_rowsK_sync(Abase + k0, lda, As, tid);
        stage_rowsK_sync(Bbase + k0, ldb, Bs, tid);
        __syncthreads();
#endif

        bf16x16 afrag[4], bfrag[2];
#pragma unroll
        for (int mt = 0; mt < 4; ++mt)
            afrag[mt] = load_afrag(As, wr * 64 + mt * 16, lane);
#pragma unroll
        for (int nt = 0; nt < 2; ++nt)
            bfrag[nt] = load_bfrag(Bs, wc * 32 + nt * 16, lane);

#pragma unroll
        for (int mt = 0; mt < 4; ++mt)
#pragma unroll
            for (int nt = 0; nt < 2; ++nt)
                acc[mt][nt] = __builtin_amdgcn_wmma_f32_16x16x32_bf16(
                    false, afrag[mt], false, bfrag[nt],
                    (short)0, acc[mt][nt], false, false);

#if HAVE_TDM
        if (wid <= 1) __builtin_amdgcn_s_wait_tensorcnt(0);
#endif
        __syncthreads();
    }

    // Epilogue. 16x16 f32 C layout: VGPR i -> M = i + 8*(lane>=16), N = lane&15.
    const int col_in = lane & 15;
    const int rhi    = (lane >> 4) * 8;
#pragma unroll
    for (int mt = 0; mt < 4; ++mt) {
#pragma unroll
        for (int nt = 0; nt < 2; ++nt) {
            const int gcol = n0 + wc * 32 + nt * 16 + col_in;
            const int gmb  = m0 + wr * 64 + mt * 16 + rhi;
#pragma unroll
            for (int i = 0; i < 8; ++i) {
                float v = acc[mt][nt][i] * scale;
                size_t idx = CT ? ((size_t)gcol * ldc + (gmb + i))
                                : ((size_t)(gmb + i) * ldc + gcol);
                if constexpr (std::is_same_v<TC, float>) C[idx] = v;
                else                                     C[idx] = (__bf16)v;
            }
        }
    }
}

// ---------------------------------------------------------------------------
// fp32 -> bf16 elementwise convert (n multiple of 1024)
// ---------------------------------------------------------------------------
__global__ __launch_bounds__(256)
void f32_to_bf16(const float* __restrict__ src, __bf16* __restrict__ dst) {
    const size_t i = ((size_t)blockIdx.x * 256 + threadIdx.x) * 4;
    float4 v = *reinterpret_cast<const float4*>(src + i);
    dst[i + 0] = (__bf16)v.x;
    dst[i + 1] = (__bf16)v.y;
    dst[i + 2] = (__bf16)v.z;
    dst[i + 3] = (__bf16)v.w;
}

// ---------------------------------------------------------------------------
// Square transpose + fp32->bf16 (dim x dim, dim multiple of 32)
// ---------------------------------------------------------------------------
__global__ __launch_bounds__(256)
void transpose_f32_bf16(const float* __restrict__ W, __bf16* __restrict__ Wt, int dim) {
    __shared__ float tile[32][33];
    const int tx = threadIdx.x & 31;
    const int ty = threadIdx.x >> 5;             // 0..7
    const int bx = blockIdx.x * 32, by = blockIdx.y * 32;
#pragma unroll
    for (int j = 0; j < 4; ++j)
        tile[ty + j * 8][tx] = W[(size_t)(by + ty + j * 8) * dim + bx + tx];
    __syncthreads();
#pragma unroll
    for (int j = 0; j < 4; ++j)
        Wt[(size_t)(bx + ty + j * 8) * dim + by + tx] = (__bf16)tile[tx][ty + j * 8];
}

// ---------------------------------------------------------------------------
// Row softmax: one 256-thread block per row of 2048 fp32, writes bf16 probs.
// ---------------------------------------------------------------------------
__global__ __launch_bounds__(256)
void softmax_rows(const float* __restrict__ S, __bf16* __restrict__ P, int ncols) {
    __shared__ float red[256];
    const int row = blockIdx.x;
    const int tid = threadIdx.x;
    const float* s = S + (size_t)row * ncols;

    float v[8];                                  // ncols == 2048 == 8*256
    float m = -INFINITY;
#pragma unroll
    for (int i = 0; i < 8; ++i) { v[i] = s[i * 256 + tid]; m = fmaxf(m, v[i]); }
    red[tid] = m;
    __syncthreads();
    for (int off = 128; off > 0; off >>= 1) {
        if (tid < off) red[tid] = fmaxf(red[tid], red[tid + off]);
        __syncthreads();
    }
    m = red[0];
    __syncthreads();

    float sum = 0.f;
#pragma unroll
    for (int i = 0; i < 8; ++i) { v[i] = expf(v[i] - m); sum += v[i]; }
    red[tid] = sum;
    __syncthreads();
    for (int off = 128; off > 0; off >>= 1) {
        if (tid < off) red[tid] += red[tid + off];
        __syncthreads();
    }
    const float inv = 1.f / red[0];

    __bf16* p = P + (size_t)row * ncols;
#pragma unroll
    for (int i = 0; i < 8; ++i) p[i * 256 + tid] = (__bf16)(v[i] * inv);
}

// ---------------------------------------------------------------------------
// Host launcher
// ---------------------------------------------------------------------------
extern "C" void kernel_launch(void* const* d_in, const int* in_sizes, int n_in,
                              void* d_out, int out_size, void* d_ws, size_t ws_size,
                              hipStream_t stream) {
    (void)in_sizes; (void)n_in; (void)out_size; (void)ws_size;

    const float* x  = (const float*)d_in[0];   // [B*S, EMB]
    const float* WQ = (const float*)d_in[1];   // [EMB, QKV]
    const float* WK = (const float*)d_in[2];
    const float* WV = (const float*)d_in[3];
    const float* W0 = (const float*)d_in[4];   // [QKV, EMB]
    float* out = (float*)d_out;                // [B*S, EMB]

    // Workspace carve (160 MB total):
    //   xb           : [16384,1024] bf16   32 MB
    //   WQt/WKt/WVt/W0t (transposed bf16)   4 x 2 MB
    //   Qb, Kb       : [16384,1024] bf16   32 MB each (Zb reuses Qb)
    //   Vt           : [1024,16384] bf16   32 MB (V transposed, per-batch cols)
    //   Sc           : [2048,2048]  fp32   16 MB
    //   Pb           : [2048,2048]  bf16    8 MB
    char* ws = (char*)d_ws;
    const size_t nTok   = (size_t)BS_ROWS * QKV_DIM;          // 16M elems
    const size_t bTok   = nTok * sizeof(__bf16);              // 32 MB
    const size_t bW     = (size_t)EMB_DIM * QKV_DIM * sizeof(__bf16); // 2 MB
    __bf16* xb  = (__bf16*)(ws);
    __bf16* WQt = (__bf16*)(ws + bTok);
    __bf16* WKt = (__bf16*)(ws + bTok + bW);
    __bf16* WVt = (__bf16*)(ws + bTok + 2 * bW);
    __bf16* W0t = (__bf16*)(ws + bTok + 3 * bW);
    __bf16* Qb  = (__bf16*)(ws + bTok + 4 * bW);
    __bf16* Kb  = (__bf16*)(ws + 2 * bTok + 4 * bW);
    __bf16* Vt  = (__bf16*)(ws + 3 * bTok + 4 * bW);
    float*  Sc  = (float*)(ws + 4 * bTok + 4 * bW);
    __bf16* Pb  = (__bf16*)(ws + 4 * bTok + 4 * bW + (size_t)SEQ * SEQ * sizeof(float));
    __bf16* Zb  = Qb;  // Q[b] is dead once S[b] is computed -> reuse

    const dim3 blk(256);
    const dim3 gT(EMB_DIM / 32, QKV_DIM / 32);     // (32,32) weight transpose
    const dim3 gProj(QKV_DIM / BN, BS_ROWS / BM);  // (8, 128)
    const dim3 gScore(SEQ / BN, SEQ / BM);         // (16, 16)
    const dim3 gPV(QKV_DIM / BN, SEQ / BM);        // (8, 16)

    // Precision prep: x -> bf16; weights -> transposed bf16 ([N][K] layout).
    f32_to_bf16<<<dim3((unsigned)(nTok / 1024)), blk, 0, stream>>>(x, xb);
    transpose_f32_bf16<<<gT, blk, 0, stream>>>(WQ, WQt, EMB_DIM);
    transpose_f32_bf16<<<gT, blk, 0, stream>>>(WK, WKt, EMB_DIM);
    transpose_f32_bf16<<<gT, blk, 0, stream>>>(WV, WVt, EMB_DIM);
    transpose_f32_bf16<<<gT, blk, 0, stream>>>(W0, W0t, QKV_DIM);

    // Projections: Q/K normal layout; V stored transposed (Vt[d][token]).
    gemm_nt_wmma<__bf16, false><<<gProj, blk, 0, stream>>>(
        xb, WQt, Qb, BS_ROWS, QKV_DIM, EMB_DIM, EMB_DIM, EMB_DIM, QKV_DIM, 1.0f);
    gemm_nt_wmma<__bf16, false><<<gProj, blk, 0, stream>>>(
        xb, WKt, Kb, BS_ROWS, QKV_DIM, EMB_DIM, EMB_DIM, EMB_DIM, QKV_DIM, 1.0f);
    gemm_nt_wmma<__bf16, true><<<gProj, blk, 0, stream>>>(
        xb, WVt, Vt, BS_ROWS, QKV_DIM, EMB_DIM, EMB_DIM, EMB_DIM, BS_ROWS, 1.0f);

    // Per-batch attention
    for (int b = 0; b < BATCH; ++b) {
        const __bf16* Qb_b = Qb + (size_t)b * SEQ * QKV_DIM;
        const __bf16* Kb_b = Kb + (size_t)b * SEQ * QKV_DIM;
        const __bf16* Vt_b = Vt + (size_t)b * SEQ;       // column slice, ld=16384
        __bf16* Zb_b = Zb + (size_t)b * SEQ * QKV_DIM;

        // S = 0.125 * Q * K^T  (K stored [token][D] == [N][K])
        gemm_nt_wmma<float, false><<<gScore, blk, 0, stream>>>(
            Qb_b, Kb_b, Sc, SEQ, SEQ, QKV_DIM, QKV_DIM, QKV_DIM, SEQ, SM_SCALE);

        softmax_rows<<<dim3(SEQ), blk, 0, stream>>>(Sc, Pb, SEQ);

        // Z = P * V  (Vt stored [d][token] == [N][K], ld = BS_ROWS)
        gemm_nt_wmma<__bf16, false><<<gPV, blk, 0, stream>>>(
            Pb, Vt_b, Zb_b, SEQ, QKV_DIM, SEQ, SEQ, BS_ROWS, QKV_DIM, 1.0f);
    }

    // out = Z * W0 (W0t stored [N][K])
    gemm_nt_wmma<float, false><<<gProj, blk, 0, stream>>>(
        Zb, W0t, out, BS_ROWS, EMB_DIM, QKV_DIM, QKV_DIM, QKV_DIM, EMB_DIM, 1.0f);
}
